// HardAttention_57423712748016
// MI455X (gfx1250) — compile-verified, matrix operand order checked by reading, not run on hardware
//
#include <hip/hip_runtime.h>
#include <hip/hip_bf16.h>

// ---------------------------------------------------------------------------
// Hard-attention glimpse RNN, 8 sequential steps.
// B=512, C=3, imsz=60, gz=8, T=8, nf=256, nh=512, classes=10.
// GEMMs on v_wmma_f32_16x16x32_bf16; rnn GEMM uses LDS-staged 64x64 tiles
// (async global->LDS via global_load_async_to_lds_b128) with 2x2 WMMA
// register blocking.
// ---------------------------------------------------------------------------

typedef __attribute__((ext_vector_type(16))) __bf16 v16bf;
typedef __attribute__((ext_vector_type(8)))  __bf16 v8bf;
typedef __attribute__((ext_vector_type(8)))  float  v8f;
typedef int v4i_vs __attribute__((vector_size(4 * sizeof(int))));

#define B_SZ    512
#define T_STEPS 8
#define NF      256
#define NH      512
#define KCAT    768   // NF + NH
#define KG      192   // C*gz*gz
#define IMSZ    60
#define NCLS    10

#if __has_builtin(__builtin_amdgcn_global_load_async_to_lds_b128)
#define HAS_ASYNC_LDS 1
#else
#define HAS_ASYNC_LDS 0
#endif

// 16B global->LDS stage (async path avoids the VGPR bounce; ASYNCcnt-tracked)
__device__ __forceinline__ void stage16(const __bf16* g, __bf16* l) {
#if HAS_ASYNC_LDS
  __builtin_amdgcn_global_load_async_to_lds_b128(
      (__attribute__((address_space(1))) v4i_vs*)g,
      (__attribute__((address_space(3))) v4i_vs*)l, 0, 0);
#else
  *(uint4*)l = *(const uint4*)g;
#endif
}

__device__ __forceinline__ void stage_wait() {
#if HAS_ASYNC_LDS
#if __has_builtin(__builtin_amdgcn_s_wait_asynccnt)
  __builtin_amdgcn_s_wait_asynccnt(0);
#else
  asm volatile("s_wait_asynccnt 0x0" ::: "memory");
#endif
#endif
  __syncthreads();
}

// 16x32 bf16 fragment layout (A row-major / B N-major), per ISA 7.12.2:
//  lane<16 : row = lane,    K = {0..7, 16..23}
//  lane>=16: row = lane-16, K = {8..15, 24..31}
// i.e. two contiguous 8-element runs at k0 and k0+16, k0 = (lane>>4)*8.
__device__ __forceinline__ v16bf load_frag(const __bf16* base, int row_base,
                                           int ld, int kb, int lane) {
  const int r  = row_base + (lane & 15);
  const int k0 = kb + ((lane >> 4) << 3);
  const __bf16* p = base + (long)r * ld + k0;
  v8bf lo = *(const v8bf*)p;
  v8bf hi = *(const v8bf*)(p + 16);
  return __builtin_shufflevector(lo, hi, 0, 1, 2, 3, 4, 5, 6, 7, 8, 9, 10, 11,
                                 12, 13, 14, 15);
}

// ---------------------------------------------------------------------------
// prep: weight conversion to bf16 (+ fused [WxhT | WhhT] pack) and zero-init.
// ---------------------------------------------------------------------------
__global__ void prep_kernel(const float* __restrict__ Wg,
                            const float* __restrict__ Wxh,
                            const float* __restrict__ Whh,
                            __bf16* __restrict__ Wg_bf,
                            __bf16* __restrict__ Bcat,
                            __bf16* __restrict__ Acat0,
                            __bf16* __restrict__ Acat1,
                            float* __restrict__ ce_acc,
                            int* __restrict__ counts) {
  const int tid    = blockIdx.x * blockDim.x + threadIdx.x;
  const int stride = gridDim.x * blockDim.x;

  for (int i = tid; i < NF * KG; i += stride)
    Wg_bf[i] = (__bf16)Wg[i];

  // Bcat[n][k]: k<NF -> Wxh[k][n] (Wxh is nf x nh), else Whh[k-NF][n]
  for (int i = tid; i < NH * KCAT; i += stride) {
    const int n = i / KCAT;
    const int k = i % KCAT;
    const float v = (k < NF) ? Wxh[k * NH + n] : Whh[(k - NF) * NH + n];
    Bcat[i] = (__bf16)v;
  }

  for (int i = tid; i < B_SZ * KCAT; i += stride) {
    Acat0[i] = (__bf16)0.0f;
    Acat1[i] = (__bf16)0.0f;
  }

  if (tid == 0) *ce_acc = 0.0f;
  if (tid < T_STEPS) counts[tid] = 0;
}

// ---------------------------------------------------------------------------
// Stage 1: feat = relu(patch . WgT + loc . Wl + bg + bl)  -> bf16 Acat[:,0:256]
// One wave per 16x16 output tile; K = 192 = 6 x 32.
// Patch rows are 8 contiguous floats in x, matching the fragment K-runs.
// ---------------------------------------------------------------------------
__global__ __launch_bounds__(32) void glimpse_feat_kernel(
    const float* __restrict__ x, const int* __restrict__ locs_t,
    const __bf16* __restrict__ Wg_bf,
    const float* __restrict__ bg, const float* __restrict__ Wl,
    const float* __restrict__ bl,
    __bf16* __restrict__ Acat_cur) {
  const int lane = threadIdx.x & 31;
  const int mb = blockIdx.x * 16;   // batch tile
  const int nb = blockIdx.y * 16;   // feature tile

  const int brow = mb + (lane & 15);
  const int r0 = locs_t[brow * 2 + 0];
  const int c0 = locs_t[brow * 2 + 1];
  const float* xb = x + (long)brow * (3 * IMSZ * IMSZ) + r0 * IMSZ + c0;

  const __bf16* wbase = Wg_bf + (long)(nb + (lane & 15)) * KG;

  v8f acc = {};
  for (int kb = 0; kb < KG; kb += 32) {
    const int k0 = kb + ((lane >> 4) << 3);   // multiple of 8 -> full patch row
    v16bf af;
#pragma unroll
    for (int half = 0; half < 2; ++half) {
      const int k = k0 + half * 16;
      const int ch = k >> 6, hh = (k & 63) >> 3;
      const float* p = xb + (ch * IMSZ + hh) * IMSZ;
#pragma unroll
      for (int j = 0; j < 8; ++j) af[half * 8 + j] = (__bf16)p[j];
    }
    const __bf16* wp = wbase + k0;
    v8bf wlo = *(const v8bf*)wp;
    v8bf whi = *(const v8bf*)(wp + 16);
    v16bf bfr = __builtin_shufflevector(wlo, whi, 0, 1, 2, 3, 4, 5, 6, 7, 8, 9,
                                        10, 11, 12, 13, 14, 15);
    acc = __builtin_amdgcn_wmma_f32_16x16x32_bf16(
        false, af, false, bfr, (short)0, acc, false, false);
  }

  const int n = nb + (lane & 15);
  const float wl0  = Wl[n];
  const float wl1  = Wl[NF + n];
  const float bsum = bg[n] + bl[n];
#pragma unroll
  for (int r = 0; r < 8; ++r) {
    const int m = mb + r + ((lane >> 4) << 3);
    const float l0 = (float)locs_t[m * 2 + 0];
    const float l1 = (float)locs_t[m * 2 + 1];
    float v = acc[r] + bsum + l0 * wl0 + l1 * wl1;
    v = v > 0.0f ? v : 0.0f;
    Acat_cur[(long)m * KCAT + n] = (__bf16)v;
  }
}

// ---------------------------------------------------------------------------
// Stage 2: hidden' = relu([feat | hidden] . [WxhT ; WhhT] + bh)
// Fused GEMM M=512, N=512, K=768. 4 waves/block, 64x64 tile:
//  - A/B 64x32 panels staged to LDS (async global->LDS)
//  - each wave: 32x32 micro-tile = 2x2 WMMA register blocking (4 wmma/k-step)
// Writes f32 hidden (classifier) and bf16 hidden into the *next* ping-pong
// buffer's columns [256:768].
// ---------------------------------------------------------------------------
__global__ __launch_bounds__(128) void rnn_step_kernel(
    const __bf16* __restrict__ Acat_cur, const __bf16* __restrict__ Bcat,
    const float* __restrict__ bh,
    float* __restrict__ hidden_f, __bf16* __restrict__ Acat_nxt) {
  __shared__ __bf16 As[64 * 32];   // rows mb..mb+63, k-slice, row-major ld=32
  __shared__ __bf16 Bs[64 * 32];   // rows nb..nb+63 of Bcat (N-major)

  const int tid  = threadIdx.x;
  const int lane = tid & 31;
  const int wv   = tid >> 5;            // 0..3
  const int mb = blockIdx.x * 64;
  const int nb = blockIdx.y * 64;
  const int wm = (wv & 1) * 32;         // wave's 32x32 micro-tile in the block
  const int wn = (wv >> 1) * 32;

  v8f acc00 = {}, acc01 = {}, acc10 = {}, acc11 = {};

  for (int kb = 0; kb < KCAT; kb += 32) {
    // ---- stage 64x32 A and B panels (256 x 16B segments each) ----
#pragma unroll
    for (int s = 0; s < 2; ++s) {
      const int seg = tid + s * 128;       // 0..255
      const int row = seg >> 2;            // 0..63
      const int off = (seg & 3) << 3;      // bf16 offset within k-slice
      stage16(Acat_cur + (long)(mb + row) * KCAT + kb + off,
              As + row * 32 + off);
      stage16(Bcat + (long)(nb + row) * KCAT + kb + off,
              Bs + row * 32 + off);
    }
    stage_wait();

    // ---- 2x2 WMMA micro-tile from LDS ----
    v16bf a0 = load_frag(As, wm, 32, 0, lane);
    v16bf a1 = load_frag(As, wm + 16, 32, 0, lane);
    v16bf b0 = load_frag(Bs, wn, 32, 0, lane);
    v16bf b1 = load_frag(Bs, wn + 16, 32, 0, lane);
    acc00 = __builtin_amdgcn_wmma_f32_16x16x32_bf16(false, a0, false, b0,
                                                    (short)0, acc00, false, false);
    acc01 = __builtin_amdgcn_wmma_f32_16x16x32_bf16(false, a0, false, b1,
                                                    (short)0, acc01, false, false);
    acc10 = __builtin_amdgcn_wmma_f32_16x16x32_bf16(false, a1, false, b0,
                                                    (short)0, acc10, false, false);
    acc11 = __builtin_amdgcn_wmma_f32_16x16x32_bf16(false, a1, false, b1,
                                                    (short)0, acc11, false, false);
    __syncthreads();
  }

  // ---- epilogue: bias + relu, write f32 hidden and bf16 next-step input ----
  const int n0 = nb + wn + (lane & 15);
  const int n1 = n0 + 16;
  const float bias0 = bh[n0];
  const float bias1 = bh[n1];
#pragma unroll
  for (int r = 0; r < 8; ++r) {
    const int m0 = mb + wm + r + ((lane >> 4) << 3);
    const int m1 = m0 + 16;
    float v;
    v = acc00[r] + bias0; v = v > 0.0f ? v : 0.0f;
    hidden_f[(long)m0 * NH + n0] = v;
    Acat_nxt[(long)m0 * KCAT + NF + n0] = (__bf16)v;
    v = acc01[r] + bias1; v = v > 0.0f ? v : 0.0f;
    hidden_f[(long)m0 * NH + n1] = v;
    Acat_nxt[(long)m0 * KCAT + NF + n1] = (__bf16)v;
    v = acc10[r] + bias0; v = v > 0.0f ? v : 0.0f;
    hidden_f[(long)m1 * NH + n0] = v;
    Acat_nxt[(long)m1 * KCAT + NF + n0] = (__bf16)v;
    v = acc11[r] + bias1; v = v > 0.0f ? v : 0.0f;
    hidden_f[(long)m1 * NH + n1] = v;
    Acat_nxt[(long)m1 * KCAT + NF + n1] = (__bf16)v;
  }
}

// ---------------------------------------------------------------------------
// Stage 3: logits, log-softmax, CE sum (atomic), accuracy count (atomic).
// N=10 is too thin for WMMA; one thread per batch row on the VALU.
// ---------------------------------------------------------------------------
__global__ __launch_bounds__(256) void classify_kernel(
    const float* __restrict__ hidden_f, const float* __restrict__ Wc,
    const float* __restrict__ bc, const int* __restrict__ y,
    float* __restrict__ ce_acc, int* __restrict__ counts, int t) {
  const int b = blockIdx.x * blockDim.x + threadIdx.x;  // 0..511

  float s[NCLS];
#pragma unroll
  for (int j = 0; j < NCLS; ++j) s[j] = bc[j];

  const float* h = hidden_f + (long)b * NH;
  for (int k = 0; k < NH; ++k) {
    const float hv = h[k];
    const float* w = Wc + (long)k * NCLS;
#pragma unroll
    for (int j = 0; j < NCLS; ++j) s[j] += hv * w[j];
  }

  float mx = s[0];
  int am = 0;
#pragma unroll
  for (int j = 1; j < NCLS; ++j)
    if (s[j] > mx) { mx = s[j]; am = j; }

  float se = 0.0f;
#pragma unroll
  for (int j = 0; j < NCLS; ++j) se += expf(s[j] - mx);
  const float lse = mx + logf(se);

  const int yy = y[b];
  const float ce = -(s[yy] - lse);

  __shared__ float red[256];
  red[threadIdx.x] = ce;
  __syncthreads();
  for (int off = 128; off > 0; off >>= 1) {
    if (threadIdx.x < off) red[threadIdx.x] += red[threadIdx.x + off];
    __syncthreads();
  }
  if (threadIdx.x == 0) atomicAdd(ce_acc, red[0]);
  if (am == yy) atomicAdd(&counts[t], 1);
}

// ---------------------------------------------------------------------------
// Finalize: d_out[0] = loss/(T*B); d_out[1+t] = counts[t]/B
// ---------------------------------------------------------------------------
__global__ void finalize_kernel(const float* __restrict__ ce_acc,
                                const int* __restrict__ counts,
                                float* __restrict__ out) {
  const int t = threadIdx.x;
  if (t == 0) out[0] = ce_acc[0] / (float)(T_STEPS * B_SZ);
  if (t < T_STEPS) out[1 + t] = (float)counts[t] / (float)B_SZ;
}

// ---------------------------------------------------------------------------
extern "C" void kernel_launch(void* const* d_in, const int* in_sizes, int n_in,
                              void* d_out, int out_size, void* d_ws, size_t ws_size,
                              hipStream_t stream) {
  (void)in_sizes; (void)n_in; (void)out_size; (void)ws_size;

  const float* x    = (const float*)d_in[0];
  const int*   y    = (const int*)  d_in[1];
  const int*   locs = (const int*)  d_in[2];
  const float* Wg   = (const float*)d_in[3];
  const float* bg   = (const float*)d_in[4];
  const float* Wl   = (const float*)d_in[5];
  const float* bl   = (const float*)d_in[6];
  const float* Wxh  = (const float*)d_in[7];
  const float* Whh  = (const float*)d_in[8];
  const float* bh   = (const float*)d_in[9];
  const float* Wc   = (const float*)d_in[10];
  const float* bc   = (const float*)d_in[11];

  char* ws = (char*)d_ws;
  // workspace layout (256B aligned offsets)
  __bf16* Wg_bf  = (__bf16*)(ws + 0);                       //  96 KB
  __bf16* Bcat   = (__bf16*)(ws + 98304);                   // 768 KB
  __bf16* Acat0  = (__bf16*)(ws + 884736);                  // 768 KB
  __bf16* Acat1  = (__bf16*)(ws + 1671168);                 // 768 KB
  float*  hidden = (float*) (ws + 2457600);                 //   1 MB
  float*  ce_acc = (float*) (ws + 3506176);
  int*    counts = (int*)   (ws + 3506176 + 256);

  prep_kernel<<<512, 256, 0, stream>>>(Wg, Wxh, Whh, Wg_bf, Bcat, Acat0, Acat1,
                                       ce_acc, counts);

  __bf16* Ab[2] = {Acat0, Acat1};
  for (int t = 0; t < T_STEPS; ++t) {
    __bf16* cur = Ab[t & 1];
    __bf16* nxt = Ab[(t + 1) & 1];
    glimpse_feat_kernel<<<dim3(B_SZ / 16, NF / 16), 32, 0, stream>>>(
        x, locs + (long)t * B_SZ * 2, Wg_bf, bg, Wl, bl, cur);
    rnn_step_kernel<<<dim3(B_SZ / 64, NH / 64), 128, 0, stream>>>(
        cur, Bcat, bh, hidden, nxt);
    classify_kernel<<<B_SZ / 256, 256, 0, stream>>>(hidden, Wc, bc, y, ce_acc,
                                                    counts, t);
  }
  finalize_kernel<<<1, 32, 0, stream>>>(ce_acc, counts, (float*)d_out);
}
